// ScannedRNN_90039694394049
// MI455X (gfx1250) — matrix operand order repaired and checked by reading, not compiled, not run on hardware
//
#include <hip/hip_runtime.h>
#include <cstdint>

// ---------------------------------------------------------------------------
// Fused LSTM scan for MI455X (gfx1250, wave32, WMMA).
//   T=512, B=256, D_IN=512, H=128, gates=4H=512.
// One workgroup (1024 threads = 32 waves) owns a 16-row batch tile and scans
// all T steps; wave w computes gate columns [16w,16w+16) via
// v_wmma_f32_16x16x32_bf16 with a bf16 hi/lo (bf16x3) split for ~fp32 accuracy.
// Wi fragments stream from L2 every step (asm barrier blocks LICM so they are
// NOT hoisted+spilled to scratch); Wh fragments (64 VGPRs) are kept resident.
// obs is streamed from HBM once (256MB) -> near the 23.3TB/s bandwidth floor.
// ---------------------------------------------------------------------------

#define T_STEPS 512
#define BATCH   256
#define DIN     512
#define HDIM    128
#define GDIM    512          // 4*H
#define MT      16           // batch rows per workgroup (WMMA M)
#define KTX     16           // DIN/32   k-tiles for x@Wi
#define KTH     4            // HDIM/32  k-tiles for h@Wh
#define NT      32           // GDIM/16  n-tiles (one per wave)
#define SROW    516          // padded row stride (floats) of gate LDS tile

typedef __attribute__((ext_vector_type(16))) __bf16   v16bf;
typedef __attribute__((ext_vector_type(2)))  __bf16   bf16x2;
typedef __attribute__((ext_vector_type(8)))  float    v8f;
typedef __attribute__((ext_vector_type(8)))  uint32_t v8u;
typedef __attribute__((ext_vector_type(4)))  uint32_t v4u;
typedef __attribute__((ext_vector_type(4)))  float    flt4;

static __device__ __forceinline__ uint32_t pack2(__bf16 a, __bf16 b) {
  bf16x2 t; t.x = a; t.y = b;
  return __builtin_bit_cast(uint32_t, t);           // k even -> [15:0], k odd -> [31:16]
}

// bf16 hi/lo split of a float pair, packed into two dwords (A/B fragment dword).
static __device__ __forceinline__ void split_pack(float a, float b, uint32_t& hi, uint32_t& lo) {
  __bf16 ah = (__bf16)a; __bf16 al = (__bf16)(a - (float)ah);
  __bf16 bh = (__bf16)b; __bf16 bl = (__bf16)(b - (float)bh);
  hi = pack2(ah, bh);
  lo = pack2(al, bl);
}

static __device__ __forceinline__ v16bf ldfrag(const uint32_t* p) {
  return __builtin_bit_cast(v16bf, *(const v8u*)p); // 32B-aligned by construction
}

static __device__ __forceinline__ v8f wmma_bf16(v16bf a, v16bf b, v8f c) {
  // (neg_a, A, neg_b, B, c_mod, C, reuse_a, reuse_b)
  return __builtin_amdgcn_wmma_f32_16x16x32_bf16(false, a, false, b, (short)0, c, false, false);
}

static __device__ __forceinline__ float sigm(float x) { return 1.f / (1.f + __expf(-x)); }
static __device__ __forceinline__ float tanh_fast(float x) {
  float e = __expf(-2.f * fabsf(x));
  float r = (1.f - e) / (1.f + e);
  return copysignf(r, x);
}

// Write one (row m, cols j,j+1) pair of h into the LDS A-fragment arrays.
// A-matrix 16x32 bf16 layout: lane = m | (khalf<<4); dwords v: kk 0..7 -> v0..3,
// kk 8..15 -> half1 v0..3, kk 16..23 -> v4..7, kk 24..31 -> half1 v4..7.
static __device__ __forceinline__ void store_hfrag(uint32_t* hi, uint32_t* lo,
                                                   int m, int j, float x0, float x1) {
  const int kt  = j >> 5;
  const int kk  = j & 31;
  const int ln  = m | (((kk >> 3) & 1) << 4);
  const int v   = ((kk & 16) ? 4 : 0) + ((kk & 7) >> 1);
  const int off = (kt * 32 + ln) * 8 + v;
  uint32_t dh, dl;
  split_pack(x0, x1, dh, dl);
  hi[off] = dh;
  lo[off] = dl;
}

// ---------------------------------------------------------------------------
// Pre-pass: split fp32 weights (K x 512 row-major) into bf16 hi/lo and swizzle
// into WMMA B-fragment order: dword index = ((kt*NT + nt)*32 + lane)*8 + v,
// where within a 32x16 tile: n = lane&15, k = 16*(v>>2) + 8*(lane>>4) + 2*(v&3) + {0,1}.
// ---------------------------------------------------------------------------
__global__ void pack_weights_kernel(const float* __restrict__ W,
                                    uint32_t* __restrict__ hiOut,
                                    uint32_t* __restrict__ loOut, int ktiles) {
  int tid = blockIdx.x * blockDim.x + threadIdx.x;
  if (tid >= ktiles * 8192) return;                 // ktiles*NT*32*8 dwords
  int v  = tid & 7;
  int ln = (tid >> 3) & 31;
  int nt = (tid >> 8) & 31;
  int kt = tid >> 13;
  int k = kt * 32 + ((v & 4) ? 16 : 0) + 8 * (ln >> 4) + 2 * (v & 3);
  int n = nt * 16 + (ln & 15);
  uint32_t hi, lo;
  split_pack(W[(size_t)k * GDIM + n], W[(size_t)(k + 1) * GDIM + n], hi, lo);
  hiOut[tid] = hi;
  loOut[tid] = lo;
}

// ---------------------------------------------------------------------------
// Persistent fused scan kernel. grid = B/MT = 16 workgroups, block = 1024.
// ---------------------------------------------------------------------------
__global__ __launch_bounds__(1024)
void lstm_scan_kernel(const float* __restrict__ c0, const float* __restrict__ h0,
                      const float* __restrict__ obs, const unsigned char* __restrict__ done,
                      const float* __restrict__ bias,
                      const uint32_t* __restrict__ WiHi, const uint32_t* __restrict__ WiLo,
                      const uint32_t* __restrict__ WhHi, const uint32_t* __restrict__ WhLo,
                      float* __restrict__ ys, float* __restrict__ cOut, float* __restrict__ hOut) {
  __shared__ uint32_t xaHi[KTX * 256];   // x A-fragments (bf16 hi), 16KB
  __shared__ uint32_t xaLo[KTX * 256];   // x A-fragments (bf16 lo), 16KB
  __shared__ uint32_t haHi[KTH * 256];   // h A-fragments (bf16 hi), 4KB
  __shared__ uint32_t haLo[KTH * 256];   // h A-fragments (bf16 lo), 4KB
  __shared__ float    Sg[MT * SROW];     // gate pre-activations, padded, ~33KB

  const int tid  = threadIdx.x;
  const int lane = tid & 31;
  const int wv   = tid >> 5;             // wave id == n-tile id (0..31)
  const int b0   = blockIdx.x * MT;

  // cooperative-load / elementwise role: row + column assignment
  const int mx = tid >> 6;               // row 0..15 (64 threads per row)
  const int k0 = (tid & 63) * 8;         // x-loader: 8 consecutive k
  const int je = (tid & 63) * 2;         // elementwise: 2 consecutive h cols

  // WMMA C/D layout: row = v + 8*(lane>>4), col = lane&15 (within n-tile)
  const int   scol = wv * 16 + (lane & 15);
  const int   srow = (lane >> 4) * 8;
  const float bsum = bias[scol];

  // Keep ONLY the small recurrent weights resident in VGPRs (8 frags = 64 regs).
  v16bf whh[KTH], whl[KTH];
  #pragma unroll
  for (int kt = 0; kt < KTH; ++kt) {
    const int fo = (kt * NT + wv) * 256 + lane * 8;
    whh[kt] = ldfrag(WhHi + fo);
    whl[kt] = ldfrag(WhLo + fo);
  }

  float creg0, creg1, hreg0, hreg1;
  {
    const float* pc = c0 + (size_t)(b0 + mx) * HDIM + je;
    creg0 = pc[0]; creg1 = pc[1];
    const float* ph = h0 + (size_t)(b0 + mx) * HDIM + je;
    hreg0 = ph[0]; hreg1 = ph[1];
    const bool keep0 = (done[(size_t)0 * BATCH + b0 + mx] == 0);  // jnp.bool_ = 1 byte
    store_hfrag(haHi, haLo, mx, je, keep0 ? hreg0 : 0.f, keep0 ? hreg1 : 0.f);
  }
  __syncthreads();

  for (int t = 0; t < T_STEPS; ++t) {
    // Optimization barrier: stops LICM from hoisting all 32 loop-invariant Wi
    // fragments out of the t-loop (which spills ~320 VGPRs to scratch). With
    // this, Wi fragments stream from the L2-resident packed array every step.
    asm volatile("" ::: "memory");

    // ---- stage x tile: 16x512 fp32 -> bf16 hi/lo A-fragments in LDS ----
    {
      const float* gp = obs + ((size_t)t * BATCH + b0 + mx) * DIN + k0;
      flt4 a  = *(const flt4*)gp;
      flt4 c2 = *(const flt4*)(gp + 4);
      if (t + 1 < T_STEPS) __builtin_prefetch(gp + (size_t)BATCH * DIN, 0, 3);
      const int kt = k0 >> 5;
      const int kk = k0 & 31;
      const int ln = mx | (((kk >> 3) & 1) << 4);
      const int vb = (kk & 16) ? 4 : 0;
      uint32_t h0w, l0w, h1w, l1w, h2w, l2w, h3w, l3w;
      split_pack(a.x,  a.y,  h0w, l0w);
      split_pack(a.z,  a.w,  h1w, l1w);
      split_pack(c2.x, c2.y, h2w, l2w);
      split_pack(c2.z, c2.w, h3w, l3w);
      v4u dh = { h0w, h1w, h2w, h3w };
      v4u dl = { l0w, l1w, l2w, l3w };
      *(v4u*)&xaHi[(kt * 32 + ln) * 8 + vb] = dh;
      *(v4u*)&xaLo[(kt * 32 + ln) * 8 + vb] = dl;
    }
    __syncthreads();

    // ---- gates = x@Wi + h@Wh via bf16x3 WMMA (fp32 accumulate) ----
    v8f acc = {};
    #pragma unroll
    for (int kt = 0; kt < KTX; ++kt) {
      const int fo = (kt * NT + wv) * 256 + lane * 8;
      v16bf bh = ldfrag(WiHi + fo);            // streamed from L2 each step
      v16bf bl = ldfrag(WiLo + fo);
      v16bf ah = ldfrag(&xaHi[(kt * 32 + lane) * 8]);
      v16bf al = ldfrag(&xaLo[(kt * 32 + lane) * 8]);
      acc = wmma_bf16(ah, bh, acc);
      acc = wmma_bf16(ah, bl, acc);
      acc = wmma_bf16(al, bh, acc);
    }
    #pragma unroll
    for (int kt = 0; kt < KTH; ++kt) {
      v16bf ah = ldfrag(&haHi[(kt * 32 + lane) * 8]);   // done[t]-masked h_{t-1}
      v16bf al = ldfrag(&haLo[(kt * 32 + lane) * 8]);
      acc = wmma_bf16(ah, whh[kt], acc);                // Wh resident in VGPRs
      acc = wmma_bf16(ah, whl[kt], acc);
      acc = wmma_bf16(al, whh[kt], acc);
    }
    #pragma unroll
    for (int v = 0; v < 8; ++v) Sg[(srow + v) * SROW + scol] = acc[v] + bsum;
    __syncthreads();

    // ---- elementwise LSTM cell; carry lives in registers ----
    {
      const float* r = &Sg[mx * SROW + je];
      const float gi0 = r[0],            gi1 = r[1];
      const float gf0 = r[HDIM],         gf1 = r[HDIM + 1];
      const float gg0 = r[2 * HDIM],     gg1 = r[2 * HDIM + 1];
      const float go0 = r[3 * HDIM],     go1 = r[3 * HDIM + 1];
      const bool keep = (done[(size_t)t * BATCH + b0 + mx] == 0);
      const float ce0 = keep ? creg0 : 0.f;
      const float ce1 = keep ? creg1 : 0.f;
      creg0 = sigm(gf0) * ce0 + sigm(gi0) * tanh_fast(gg0);
      creg1 = sigm(gf1) * ce1 + sigm(gi1) * tanh_fast(gg1);
      hreg0 = sigm(go0) * tanh_fast(creg0);
      hreg1 = sigm(go1) * tanh_fast(creg1);
      float* py = ys + ((size_t)t * BATCH + b0 + mx) * HDIM + je;
      py[0] = hreg0; py[1] = hreg1;
      // done[t+1] is a static input: apply next step's reset at write time.
      const bool keepN = (t + 1 < T_STEPS) ? (done[(size_t)(t + 1) * BATCH + b0 + mx] == 0) : true;
      store_hfrag(haHi, haLo, mx, je, keepN ? hreg0 : 0.f, keepN ? hreg1 : 0.f);
    }
    __syncthreads();
  }

  // final carries (unmasked, post step T-1)
  {
    float* pc = cOut + (size_t)(b0 + mx) * HDIM + je;
    pc[0] = creg0; pc[1] = creg1;
    float* ph = hOut + (size_t)(b0 + mx) * HDIM + je;
    ph[0] = hreg0; ph[1] = hreg1;
  }
}

// ---------------------------------------------------------------------------
extern "C" void kernel_launch(void* const* d_in, const int* in_sizes, int n_in,
                              void* d_out, int out_size, void* d_ws, size_t ws_size,
                              hipStream_t stream) {
  (void)in_sizes; (void)n_in; (void)out_size; (void)ws_size;

  const float*         c0   = (const float*)d_in[0];
  const float*         h0   = (const float*)d_in[1];
  const float*         obs  = (const float*)d_in[2];
  const unsigned char* done = (const unsigned char*)d_in[3];  // jnp.bool_ -> 1 byte/elem
  const float*         Wi   = (const float*)d_in[4];
  const float*         Wh   = (const float*)d_in[5];
  const float*         bias = (const float*)d_in[6];

  // scratch: swizzled bf16 hi/lo weight fragments (~1.25 MB, L2-resident)
  uint32_t* ws   = (uint32_t*)d_ws;
  uint32_t* WiHi = ws;
  uint32_t* WiLo = WiHi + KTX * NT * 256;  // 131072 dwords
  uint32_t* WhHi = WiLo + KTX * NT * 256;
  uint32_t* WhLo = WhHi + KTH * NT * 256;  // 32768 dwords

  pack_weights_kernel<<<(KTX * 8192 + 255) / 256, 256, 0, stream>>>(Wi, WiHi, WiLo, KTX);
  pack_weights_kernel<<<(KTH * 8192 + 255) / 256, 256, 0, stream>>>(Wh, WhHi, WhLo, KTH);

  float* ys   = (float*)d_out;                          // [T,B,H]
  float* cOut = ys + (size_t)T_STEPS * BATCH * HDIM;    // [B,H]
  float* hOut = cOut + (size_t)BATCH * HDIM;            // [B,H]

  lstm_scan_kernel<<<BATCH / MT, 1024, 0, stream>>>(c0, h0, obs, done, bias,
                                                    WiHi, WiLo, WhHi, WhLo,
                                                    ys, cOut, hOut);
}
